// CacheOffloadedTransformer_76441827934975
// MI455X (gfx1250) — matrix operand order repaired and checked by reading, not compiled, not run on hardware
//
#include <hip/hip_runtime.h>

typedef __attribute__((ext_vector_type(16))) _Float16 v16h;
typedef __attribute__((ext_vector_type(8)))  float    v8f;
typedef __attribute__((ext_vector_type(2)))  float    v2f;
typedef __attribute__((ext_vector_type(4)))  int      v4i;

#define AS1 __attribute__((address_space(1)))
#define AS3 __attribute__((address_space(3)))

#if defined(__has_builtin)
#  if __has_builtin(__builtin_amdgcn_wmma_f32_16x16x4_f32)
#    define HAVE_WMMA_F32_16X16X4 1
#  endif
#  if __has_builtin(__builtin_amdgcn_global_load_async_to_lds_b128) && \
      __has_builtin(__builtin_amdgcn_s_wait_asynccnt)
#    define HAVE_ASYNC_LDS 1
#  endif
#endif

#ifdef HAVE_WMMA_F32_16X16X4
#pragma message("CDNA5: using V_WMMA_F32_16X16X4_F32 (fp32) path")
#else
#pragma message("CDNA5: using V_WMMA_F32_16X16X32_F16 fallback path")
#endif
#ifdef HAVE_ASYNC_LDS
#pragma message("CDNA5: async global->LDS (ASYNCcnt) path enabled")
#else
#pragma message("CDNA5: async global->LDS builtins not found; direct loads")
#endif

namespace cot {

constexpr int B   = 2;
constexpr int S   = 2048;
constexpr int HID = 2048;
constexpr int NH  = 16;
constexpr int L   = 8;
constexpr int V   = 32000;
constexpr int D   = 128;           // head dim
constexpr int OFF = 1024;          // offload rows
constexpr int R   = 1025;          // resident rows
constexpr int P   = 11272;         // physical heap rows
constexpr int H3  = 3 * HID;       // 6144
constexpr int H4  = 4 * HID;       // 8192
constexpr int KLEN   = 2049;       // keys attended per layer
constexpr int NSPLIT = 16;         // attention key splits (flash-decode)
constexpr int ACHUNK = (KLEN + NSPLIT - 1) / NSPLIT;  // 129

// ---- workspace layout (floats) ----
constexpr size_t WS_X      = 0;                                   // [B][HID]
constexpr size_t WS_XN     = WS_X    + (size_t)B * HID;           // [B][HID]
constexpr size_t WS_QKV    = WS_XN   + (size_t)B * HID;           // [B][3H]
constexpr size_t WS_H1     = WS_QKV  + (size_t)B * H3;            // [B][4H]
constexpr size_t WS_NEWK   = WS_H1   + (size_t)B * H4;            // [L][B][NH][D]
constexpr size_t WS_NEWV   = WS_NEWK + (size_t)L * B * NH * D;
constexpr size_t WS_LOGITS = WS_NEWV + (size_t)L * B * NH * D;    // [B][V]
constexpr size_t WS_PART   = WS_LOGITS + (size_t)B * V;           // max nsplit*B*N = 128000
constexpr size_t WS_ATTNP  = WS_PART + 128000;                    // [B*NH][NSPLIT][D+2]

__device__ __forceinline__ unsigned lds_off32(const void* p) {
  // AS3 offset == low 32 bits of the flat address
  return (unsigned)(unsigned long long)p;
}

// ---------------------------------------------------------------------------
// x = x_in + sinusoidal PE(current_pos)
// ---------------------------------------------------------------------------
__global__ __launch_bounds__(256) void pe_add_kernel(
    const float* __restrict__ x_in, const int* __restrict__ pos_p,
    float* __restrict__ x)
{
  int i = blockIdx.x * 256 + threadIdx.x;
  if (i >= B * HID) return;
  int d    = i % HID;
  int half = d >> 1;
  float pos = (float)pos_p[0];
  float div = __expf(-logf(10000.0f) * (2.0f * (float)half) / (float)HID);
  float ang = pos * div;
  float pe  = (d & 1) ? cosf(ang) : sinf(ang);
  x[i] = x_in[i] + pe;
}

// ---------------------------------------------------------------------------
// LayerNorm over HID for each batch row (grid = B, block = 256)
// ---------------------------------------------------------------------------
__global__ __launch_bounds__(256) void ln_kernel(
    const float* __restrict__ x, const float* __restrict__ g,
    const float* __restrict__ bb, float* __restrict__ xn)
{
  __shared__ float red[256];
  int b = blockIdx.x, tid = threadIdx.x;
  const float* xb = x + (size_t)b * HID;

  float s = 0.f;
  for (int i = tid; i < HID; i += 256) s += xb[i];
  red[tid] = s; __syncthreads();
  for (int st = 128; st > 0; st >>= 1) {
    if (tid < st) red[tid] += red[tid + st];
    __syncthreads();
  }
  float mean = red[0] * (1.0f / HID);
  __syncthreads();

  float v = 0.f;
  for (int i = tid; i < HID; i += 256) { float dd = xb[i] - mean; v += dd * dd; }
  red[tid] = v; __syncthreads();
  for (int st = 128; st > 0; st >>= 1) {
    if (tid < st) red[tid] += red[tid + st];
    __syncthreads();
  }
  float inv = rsqrtf(red[0] * (1.0f / HID) + 1e-5f);

  for (int i = tid; i < HID; i += 256)
    xn[(size_t)b * HID + i] = (xb[i] - mean) * inv * g[i] + bb[i];
}

// ---------------------------------------------------------------------------
// Split-K GEMV via WMMA: partial[split][n][row] = dot(W[row, kb:ke], x[n, kb:ke])
// W: [N,K] row-major fp32. One wave per 16-row tile; grid.y = K-splits.
// A operand = 16 weight rows (streamed at 100% utilization), B = x^T.
// Inner loop explicitly unrolled: 8 WMMAs per prefetched 128B k-stripe.
// ---------------------------------------------------------------------------
__global__ __launch_bounds__(256) void gemv_wmma_kernel(
    const float* __restrict__ W, const float* __restrict__ xin,
    float* __restrict__ part, int N, int K, int kchunk)
{
  int lane = threadIdx.x & 31;
  int wv   = threadIdx.x >> 5;
  int row0 = (blockIdx.x * 8 + wv) * 16;
  if (row0 >= N) return;                 // wave-uniform; EXEC stays all-ones

  int split = blockIdx.y;
  int kb = split * kchunk;
  int ke = kb + kchunk;                  // kchunk is a multiple of 32

  int m  = lane & 15;                    // A row in tile / B column (batch id)
  int kh = lane >> 4;                    // K-half select per ISA layout
  v8f c = {};                            // 16x16 f32 accumulator (8 VGPRs)

#ifdef HAVE_WMMA_F32_16X16X4
  // --- fp32 path: V_WMMA_F32_16X16X4_F32 ---
  const float* wrow = W + (size_t)(row0 + m) * K + 2 * kh;          // A lanes
  const float* xrow = xin + (size_t)(m < 2 ? m : 0) * K + 2 * kh;   // B lanes
  for (int k0 = kb; k0 < ke; k0 += 32) {
    __builtin_prefetch(wrow + k0 + 2048, 0, 3);   // global_prefetch_b8
#pragma unroll
    for (int ku = 0; ku < 32; ku += 4) {
      int k = k0 + ku;
      v2f a;  a.x = wrow[k];  a.y = wrow[k + 1];
      v2f bm = {0.f, 0.f};
      if (m < 2) { bm.x = xrow[k]; bm.y = xrow[k + 1]; }
      c = __builtin_amdgcn_wmma_f32_16x16x4_f32(
          false, a, false, bm, (short)0, c, false, false);
    }
  }
#else
  // --- fallback: V_WMMA_F32_16X16X32_F16 (convert-on-load, f32 accumulate) ---
  const float* wrow = W + (size_t)(row0 + m) * K + 16 * kh;
  const float* xrow = xin + (size_t)(m < 2 ? m : 0) * K + 16 * kh;
  for (int k0 = kb; k0 < ke; k0 += 32) {
    __builtin_prefetch(wrow + k0 + 2048, 0, 3);
    const float* wa = wrow + k0;
    v16h a, bm;
#pragma unroll
    for (int e = 0; e < 8; ++e) {
      a[e]     = (_Float16)wa[e];
      a[e + 8] = (_Float16)wa[e + 8];
    }
    if (m < 2) {
      const float* xb = xrow + k0;
#pragma unroll
      for (int e = 0; e < 16; ++e) bm[e] = (_Float16)xb[e];
    } else {
#pragma unroll
      for (int e = 0; e < 16; ++e) bm[e] = (_Float16)0.f;
    }
    c = __builtin_amdgcn_wmma_f32_16x16x32_f16(
        false, a, false, bm, (short)0, c, false, false);
  }
#endif

  // D layout: VGPR r, lanes 0-15 -> M=r, lanes 16-31 -> M=8+r; N-col = lane&15
  if (m < 2) {
    float* o = part + ((size_t)split * B + m) * N + row0 + 8 * kh;
#pragma unroll
    for (int r = 0; r < 8; ++r) o[r] = c[r];
  }
}

// Reduce split-K partials. op: 0 = store, 1 = GELU(store), 2 = accumulate.
__global__ __launch_bounds__(256) void gemv_reduce_kernel(
    const float* __restrict__ part, float* __restrict__ out,
    int N, int nsplit, int op)
{
  int i = blockIdx.x * 256 + threadIdx.x;
  if (i >= B * N) return;
  int b = i / N, row = i % N;
  float v = 0.f;
  for (int s = 0; s < nsplit; ++s) v += part[((size_t)s * B + b) * N + row];
  if (op == 1) v = 0.5f * v * (1.0f + erff(v * 0.7071067811865476f));
  float* o = out + (size_t)b * N + row;
  if (op == 2) *o += v; else *o = v;
}

// ---------------------------------------------------------------------------
// Extract this layer's new K/V rows from the qkv vector
// ---------------------------------------------------------------------------
__global__ __launch_bounds__(256) void split_kv_kernel(
    const float* __restrict__ qkv, float* __restrict__ nK, float* __restrict__ nV)
{
  int i = blockIdx.x * 256 + threadIdx.x;
  if (i >= B * NH * D) return;
  int b = i / (NH * D);
  int r = i % (NH * D);
  int h = r / D, d = r % D;
  const float* base = qkv + (size_t)b * H3 + (size_t)h * 3 * D;
  nK[i] = base[D + d];
  nV[i] = base[2 * D + d];
}

// ---------------------------------------------------------------------------
// Flash-decode attention pass 1. grid = (B*NH, NSPLIT), block = 256 (8 waves).
// Each wave owns a strided slice of the block's key chunk; K/V rows are
// double-buffered through LDS with GLOBAL_LOAD_ASYNC_TO_LDS_B128 (ASYNCcnt),
// so key j+8 streams in while key j is scored. Waves combine in LDS and emit
// one (m, s, acc[D]) partial. New KV rows come from workspace (inputs never
// mutated); wpos algebra: layer l's scatter write lands at resident-window
// offset 1023 of layer l+2; layer 0 reads its own appended row at heap row S.
// ---------------------------------------------------------------------------
__global__ __launch_bounds__(256) void attn_kernel(
    const float* __restrict__ qkv,
    const float* __restrict__ k_heap, const float* __restrict__ v_heap,
    const float* __restrict__ k_off,  const float* __restrict__ v_off,
    const float* __restrict__ newK,   const float* __restrict__ newV,
    float* __restrict__ attnP, int l)
{
  constexpr int NW = 8;
  __shared__ float sm[NW], ss[NW], sacc[NW][D];
#ifdef HAVE_ASYNC_LDS
  __shared__ float stage[NW][2][2][D];   // [wave][buf][K|V][dim]  16 KB
#endif

  int bh = blockIdx.x;
  int split = blockIdx.y;
  int b = bh / NH, h = bh % NH;
  int lane = threadIdx.x & 31;
  int wv   = threadIdx.x >> 5;

  int jb = split * ACHUNK;
  int je = jb + ACHUNK < KLEN ? jb + ACHUNK : KLEN;

  const float4 q4 = *(const float4*)(qkv + (size_t)b * H3 + (size_t)h * 3 * D + lane * 4);
  const float scale = 0.08838834764831845f;  // 1/sqrt(128)

  const size_t heap_bh = ((size_t)b * NH + h) * (size_t)P;
  const size_t off_bh  = (l >= 1)
      ? ((((size_t)(l - 1) * B + b) * NH + h) * (size_t)OFF) : 0;
  const size_t sub_l   = (l >= 2) ? (size_t)(l - 2) : 0;
  const float* nK0 = newK + ((size_t)b * NH + h) * D;       // layer 0's new row
  const float* nV0 = newV + ((size_t)b * NH + h) * D;
  const float* nKs = newK + (sub_l * B * NH + (size_t)b * NH + h) * D;
  const float* nVs = newV + (sub_l * B * NH + (size_t)b * NH + h) * D;

  auto kv_addr = [&](int j, const float*& kp, const float*& vp) {
    if (l == 0) {
      if (j < S) { kp = k_heap + (heap_bh + j) * D; vp = v_heap + (heap_bh + j) * D; }
      else       { kp = nK0;                         vp = nV0; }
    } else {
      if (j < R) {
        if (l >= 2 && j == R - 2) { kp = nKs; vp = nVs; }   // window offset 1023
        else {
          size_t row = heap_bh + (size_t)(l - 1) * R + j;
          kp = k_heap + row * D; vp = v_heap + row * D;
        }
      } else {
        size_t row = off_bh + (size_t)(j - R);
        kp = k_off + row * D; vp = v_off + row * D;
      }
    }
  };

  float mrun = -__builtin_inff();
  float ssum = 0.f;
  float4 acc = {0.f, 0.f, 0.f, 0.f};

#ifdef HAVE_ASYNC_LDS
  // --- async double-buffered KV stream through LDS ---
  auto issue = [&](int j, int buf) {
    const float *kp, *vp;
    kv_addr(j, kp, vp);
    __builtin_amdgcn_global_load_async_to_lds_b128(
        (AS1 v4i*)(unsigned long long)(kp + lane * 4),
        (AS3 v4i*)lds_off32(&stage[wv][buf][0][lane * 4]), 0, 0);
    __builtin_amdgcn_global_load_async_to_lds_b128(
        (AS1 v4i*)(unsigned long long)(vp + lane * 4),
        (AS3 v4i*)lds_off32(&stage[wv][buf][1][lane * 4]), 0, 0);
  };
  int j = jb + wv;
  int buf = 0;
  if (j < je) issue(j, 0);
  for (; j < je; j += NW) {
    if (j + NW < je) issue(j + NW, buf ^ 1);
    __builtin_amdgcn_s_wait_asynccnt(2);   // current buffer landed; next in flight
    float4 k4 = *(const float4*)(&stage[wv][buf][0][lane * 4]);
    float4 v4 = *(const float4*)(&stage[wv][buf][1][lane * 4]);
    float p = q4.x * k4.x + q4.y * k4.y + q4.z * k4.z + q4.w * k4.w;
#pragma unroll
    for (int o = 16; o > 0; o >>= 1) p += __shfl_xor(p, o, 32);
    float sj   = p * scale;
    float mn   = fmaxf(mrun, sj);
    float corr = __expf(mrun - mn);   // exp(-inf) = 0 on first iteration
    float e    = __expf(sj - mn);
    acc.x = acc.x * corr + e * v4.x;
    acc.y = acc.y * corr + e * v4.y;
    acc.z = acc.z * corr + e * v4.z;
    acc.w = acc.w * corr + e * v4.w;
    ssum  = ssum * corr + e;
    mrun  = mn;
    buf ^= 1;
  }
  __builtin_amdgcn_s_wait_asynccnt(0);
#else
  for (int j = jb + wv; j < je; j += NW) {
    const float *kp, *vp;
    kv_addr(j, kp, vp);
    float4 k4 = *(const float4*)(kp + lane * 4);
    float p = q4.x * k4.x + q4.y * k4.y + q4.z * k4.z + q4.w * k4.w;
#pragma unroll
    for (int o = 16; o > 0; o >>= 1) p += __shfl_xor(p, o, 32);
    float sj   = p * scale;
    float mn   = fmaxf(mrun, sj);
    float corr = __expf(mrun - mn);
    float e    = __expf(sj - mn);
    float4 v4  = *(const float4*)(vp + lane * 4);
    acc.x = acc.x * corr + e * v4.x;
    acc.y = acc.y * corr + e * v4.y;
    acc.z = acc.z * corr + e * v4.z;
    acc.w = acc.w * corr + e * v4.w;
    ssum  = ssum * corr + e;
    mrun  = mn;
  }
#endif

  if (lane == 0) { sm[wv] = mrun; ss[wv] = ssum; }
  *(float4*)(&sacc[wv][lane * 4]) = acc;
  __syncthreads();

  // combine the 8 waves, emit one partial for this (bh, split)
  float* Pp = attnP + ((size_t)bh * NSPLIT + split) * (D + 2);
  if (threadIdx.x < D) {
    int d = threadIdx.x;
    float M = -__builtin_inff();
#pragma unroll
    for (int w = 0; w < NW; ++w) M = fmaxf(M, sm[w]);
    float tot = 0.f, o = 0.f;
#pragma unroll
    for (int w = 0; w < NW; ++w) {
      float cw = __expf(sm[w] - M);
      tot += ss[w] * cw;
      o   += sacc[w][d] * cw;
    }
    if (d == 0) { Pp[0] = M; Pp[1] = tot; }
    Pp[2 + d] = o;
  }
}

// Attention pass 2: combine NSPLIT partials, add to residual x.
__global__ __launch_bounds__(128) void attn_combine_kernel(
    const float* __restrict__ attnP, float* __restrict__ x)
{
  int bh = blockIdx.x;
  int b = bh / NH, h = bh % NH;
  int d = threadIdx.x;

  float M = -__builtin_inff();
#pragma unroll
  for (int s = 0; s < NSPLIT; ++s)
    M = fmaxf(M, attnP[((size_t)bh * NSPLIT + s) * (D + 2)]);
  float tot = 0.f, o = 0.f;
#pragma unroll
  for (int s = 0; s < NSPLIT; ++s) {
    const float* Pp = attnP + ((size_t)bh * NSPLIT + s) * (D + 2);
    float cw = __expf(Pp[0] - M);
    tot += Pp[1] * cw;
    o   += Pp[2 + d] * cw;
  }
  x[(size_t)b * HID + (size_t)h * D + d] += o / tot;
}

// ---------------------------------------------------------------------------
// Final softmax over V per batch (grid = B, block = 256)
// ---------------------------------------------------------------------------
__global__ __launch_bounds__(256) void softmax_kernel(
    const float* __restrict__ logits, float* __restrict__ out)
{
  __shared__ float red[256];
  int b = blockIdx.x, tid = threadIdx.x;
  const float* lb = logits + (size_t)b * V;

  float mx = -__builtin_inff();
  for (int i = tid; i < V; i += 256) mx = fmaxf(mx, lb[i]);
  red[tid] = mx; __syncthreads();
  for (int st = 128; st > 0; st >>= 1) {
    if (tid < st) red[tid] = fmaxf(red[tid], red[tid + st]);
    __syncthreads();
  }
  float M = red[0]; __syncthreads();

  float s = 0.f;
  for (int i = tid; i < V; i += 256) s += __expf(lb[i] - M);
  red[tid] = s; __syncthreads();
  for (int st = 128; st > 0; st >>= 1) {
    if (tid < st) red[tid] += red[tid + st];
    __syncthreads();
  }
  float inv = 1.0f / red[0];

  for (int i = tid; i < V; i += 256)
    out[(size_t)b * V + i] = __expf(lb[i] - M) * inv;
}

} // namespace cot

// ---------------------------------------------------------------------------
extern "C" void kernel_launch(void* const* d_in, const int* in_sizes, int n_in,
                              void* d_out, int out_size, void* d_ws, size_t ws_size,
                              hipStream_t stream) {
  using namespace cot;
  const float* x_in   = (const float*)d_in[0];
  const float* qkv_w  = (const float*)d_in[1];
  const float* ffn1_w = (const float*)d_in[2];
  const float* ffn2_w = (const float*)d_in[3];
  const float* out_w  = (const float*)d_in[4];
  const float* ln_g   = (const float*)d_in[5];
  const float* ln_b   = (const float*)d_in[6];
  const float* k_heap = (const float*)d_in[7];
  const float* v_heap = (const float*)d_in[8];
  const float* k_off  = (const float*)d_in[9];
  const float* v_off  = (const float*)d_in[10];
  const int*   posp   = (const int*)d_in[11];

  float* ws     = (float*)d_ws;
  float* x      = ws + WS_X;
  float* xn     = ws + WS_XN;
  float* qkv    = ws + WS_QKV;
  float* h1     = ws + WS_H1;
  float* newK   = ws + WS_NEWK;
  float* newV   = ws + WS_NEWV;
  float* logits = ws + WS_LOGITS;
  float* part   = ws + WS_PART;
  float* attnP  = ws + WS_ATTNP;

  auto gemv = [&](const float* W, const float* xin, float* out,
                  int N, int K, int nsplit, int op) {
    dim3 grid((N + 127) / 128, nsplit);
    gemv_wmma_kernel<<<grid, 256, 0, stream>>>(W, xin, part, N, K, K / nsplit);
    gemv_reduce_kernel<<<(B * N + 255) / 256, 256, 0, stream>>>(
        part, out, N, nsplit, op);
  };

  pe_add_kernel<<<(B * HID + 255) / 256, 256, 0, stream>>>(x_in, posp, x);

  for (int l = 0; l < L; ++l) {
    ln_kernel<<<B, 256, 0, stream>>>(x, ln_g, ln_b, xn);

    gemv(qkv_w + (size_t)l * H3 * HID, xn, qkv, H3, HID, /*nsplit=*/4, /*op=*/0);

    split_kv_kernel<<<(B * NH * D + 255) / 256, 256, 0, stream>>>(
        qkv, newK + (size_t)l * B * NH * D, newV + (size_t)l * B * NH * D);

    attn_kernel<<<dim3(B * NH, NSPLIT), 256, 0, stream>>>(
        qkv, k_heap, v_heap, k_off, v_off, newK, newV, attnP, l);
    attn_combine_kernel<<<B * NH, 128, 0, stream>>>(attnP, x);

    ln_kernel<<<B, 256, 0, stream>>>(x, ln_g, ln_b, xn);

    gemv(ffn1_w + (size_t)l * H4 * HID, xn, h1, H4, HID, /*nsplit=*/4, /*op=*/1);
    gemv(ffn2_w + (size_t)l * HID * H4, h1, x, HID, H4, /*nsplit=*/16, /*op=*/2);
  }

  ln_kernel<<<B, 256, 0, stream>>>(x, ln_g, ln_b, xn);
  gemv(out_w, xn, logits, V, HID, /*nsplit=*/2, /*op=*/0);
  softmax_kernel<<<B, 256, 0, stream>>>(logits, (float*)d_out);
}
